// Autoformer_71287867179267
// MI455X (gfx1250) — compile-verified, matrix operand order checked by reading, not run on hardware
//
#include <hip/hip_runtime.h>
#include <math.h>

// ---------------------------------------------------------------------------
// CDNA5 (gfx1250) WMMA types
// ---------------------------------------------------------------------------
typedef __attribute__((ext_vector_type(16))) __bf16 v16bf;
typedef __attribute__((ext_vector_type(8)))  __bf16 v8bf;
typedef __attribute__((ext_vector_type(8)))  float  v8f;

// LDS tile row length: 32 payload + 8 pad (80 B row stride = 20 banks,
// keeps 16B alignment for b128 fragment loads, breaks 4-way conflicts).
#define LDSROW 40

__device__ __forceinline__ __bf16 f2bf(float f) {
  unsigned u = __builtin_bit_cast(unsigned, f);
  unsigned r = u + 0x7FFFu + ((u >> 16) & 1u);   // round-to-nearest-even
  unsigned short h = (unsigned short)(r >> 16);
  return __builtin_bit_cast(__bf16, h);
}

__device__ __forceinline__ float act_apply(float v, int act) {
  if (act == 1) {                     // GELU (tanh approx, jax default)
    const float c = 0.7978845608028654f;
    return 0.5f * v * (1.f + tanhf(c * (v + 0.044715f * v * v * v)));
  } else if (act == 2) {              // ELU
    return v > 0.f ? v : (expf(v) - 1.f);
  }
  return v;
}

// ISA 16-bit fragment layout (A 16x32 / B as 32x16 col-view): lanes 0-15 hold
// K = {0..7, 16..23}, lanes 16-31 hold K = {8..15, 24..31}. With rows stored
// contiguously in LDS that is exactly two 16-byte runs -> 2x ds_load_b128.
__device__ __forceinline__ v16bf load_frag(const __bf16* rowBase, int khalf) {
  v8bf lo = *(const v8bf*)(rowBase + khalf);
  v8bf hi = *(const v8bf*)(rowBase + 16 + khalf);
  return __builtin_shufflevector(lo, hi, 0, 1, 2, 3, 4, 5, 6, 7,
                                 8, 9, 10, 11, 12, 13, 14, 15);
}

// ---------------------------------------------------------------------------
// Generic WMMA GEMM:  C[M,N] = act( A[M,K] * W[K,N] + bias )
// amode==1: A is a circular-pad-1 conv3 im2col view of x[(B,L,Cin)],
//           logical K = 3*Cin, k -> (j = k/Cin, c = k%Cin), row -> (b,t),
//           element = x[b, (t-1+j) mod L, c].
// Requires K % 32 == 0. Block = 128 threads (4 waves), 64x64 tile.
// ---------------------------------------------------------------------------
__global__ void __launch_bounds__(128)
gemm_wmma(const float* __restrict__ A, const float* __restrict__ W,
          const float* __restrict__ bias, float* __restrict__ C,
          int M, int N, int K, int act, int amode, int L, int Cin)
{
  __shared__ __align__(16) __bf16 As[64][LDSROW];   // [m][k]
  __shared__ __align__(16) __bf16 Bs[64][LDSROW];   // [n][k]  (transposed W tile)
  const int tid   = threadIdx.x;
  const int row0  = blockIdx.y * 64;
  const int n0    = blockIdx.x * 64;
  const int wave  = tid >> 5;
  const int lane  = tid & 31;
  const int wm    = (wave >> 1) * 32;
  const int wn    = (wave & 1) * 32;
  const int lrow  = lane & 15;
  const int khalf = (lane >> 4) * 8;

  v8f acc[2][2];
#pragma unroll
  for (int i = 0; i < 2; ++i)
#pragma unroll
    for (int j = 0; j < 2; ++j)
#pragma unroll
      for (int e = 0; e < 8; ++e) acc[i][j][e] = 0.f;

  for (int k0 = 0; k0 < K; k0 += 32) {
    for (int i = tid; i < 64 * 32; i += 128) {
      int r = i >> 5, c = i & 31;                  // coalesced in k
      int gr = row0 + r, kk = k0 + c;
      float v = 0.f;
      if (gr < M) {
        if (amode == 0) {
          v = A[(size_t)gr * K + kk];
        } else {
          int b = gr / L, t = gr - b * L;
          int j = kk / Cin, ci = kk - j * Cin;
          int tt = t - 1 + j;
          tt += (tt < 0) ? L : 0;
          tt -= (tt >= L) ? L : 0;
          v = A[((size_t)b * L + tt) * (size_t)Cin + ci];
        }
      }
      As[r][c] = f2bf(v);
    }
    for (int i = tid; i < 32 * 64; i += 128) {
      int r = i >> 6, c = i & 63;                  // coalesced in n
      int n = n0 + c;
      float v = (n < N) ? W[(size_t)(k0 + r) * N + n] : 0.f;
      Bs[c][r] = f2bf(v);                          // store transposed
    }
    __syncthreads();

    v16bf afr[2], bfr[2];
#pragma unroll
    for (int fm = 0; fm < 2; ++fm)
      afr[fm] = load_frag(&As[wm + fm * 16 + lrow][0], khalf);
#pragma unroll
    for (int fn = 0; fn < 2; ++fn)
      bfr[fn] = load_frag(&Bs[wn + fn * 16 + lrow][0], khalf);
#pragma unroll
    for (int fm = 0; fm < 2; ++fm)
#pragma unroll
      for (int fn = 0; fn < 2; ++fn)
        acc[fm][fn] = __builtin_amdgcn_wmma_f32_16x16x32_bf16(
            false, afr[fm], false, bfr[fn], (short)0, acc[fm][fn], false, false);
    __syncthreads();
  }

#pragma unroll
  for (int fm = 0; fm < 2; ++fm)
#pragma unroll
    for (int fn = 0; fn < 2; ++fn)
#pragma unroll
      for (int e = 0; e < 8; ++e) {
        int m = row0 + wm + fm * 16 + e + ((lane >> 4) << 3);
        int n = n0 + wn + fn * 16 + (lane & 15);
        if (m < M && n < N) {
          float v = acc[fm][fn][e];
          if (bias) v += bias[n];
          C[(size_t)m * N + n] = act_apply(v, act);
        }
      }
}

// ---------------------------------------------------------------------------
// Circular autocorrelation mean via WMMA Gram-diagonal reduction.
// mean_value[b,tau] = (1/256) * sum_s <Q[b,(s+tau)%L,:], K[b,s,:]>
// Each block computes a 64x64 tile of Q*K^T for one batch, folds it into the
// 127 diagonals it touches (LDS ds_add_f32), then atomic-adds corr[b,tau].
// B = K^T, so the natural row-major (s,d) store order is already the
// fragment-contiguous [n][k] layout -- conflict-free stores + b128 loads.
// ---------------------------------------------------------------------------
__global__ void __launch_bounds__(128)
corr_wmma(const float* __restrict__ Q, const float* __restrict__ Kmat,
          float* __restrict__ corr, int L, int D, int qBatchRows, int kBatchRows)
{
  __shared__ __align__(16) __bf16 As[64][LDSROW];  // [t][d]
  __shared__ __align__(16) __bf16 Bs[64][LDSROW];  // [s][d]
  __shared__ float diag[127];
  const int tid   = threadIdx.x;
  const int b     = blockIdx.z;
  const int t0    = blockIdx.x * 64;
  const int s0    = blockIdx.y * 64;
  const int wave  = tid >> 5;
  const int lane  = tid & 31;
  const int wm    = (wave >> 1) * 32;
  const int wn    = (wave & 1) * 32;
  const int lrow  = lane & 15;
  const int khalf = (lane >> 4) * 8;

  for (int i = tid; i < 127; i += 128) diag[i] = 0.f;

  v8f acc[2][2];
#pragma unroll
  for (int i = 0; i < 2; ++i)
#pragma unroll
    for (int j = 0; j < 2; ++j)
#pragma unroll
      for (int e = 0; e < 8; ++e) acc[i][j][e] = 0.f;

  const float* Qb = Q    + (size_t)b * qBatchRows * D;
  const float* Kb = Kmat + (size_t)b * kBatchRows * D;

  for (int k0 = 0; k0 < D; k0 += 32) {
    for (int i = tid; i < 64 * 32; i += 128) {
      int r = i >> 5, c = i & 31;
      As[r][c] = f2bf(Qb[(size_t)(t0 + r) * D + k0 + c]);
    }
    for (int i = tid; i < 64 * 32; i += 128) {
      int s = i >> 5, dk = i & 31;
      Bs[s][dk] = f2bf(Kb[(size_t)(s0 + s) * D + k0 + dk]);
    }
    __syncthreads();

    v16bf afr[2], bfr[2];
#pragma unroll
    for (int fm = 0; fm < 2; ++fm)
      afr[fm] = load_frag(&As[wm + fm * 16 + lrow][0], khalf);
#pragma unroll
    for (int fn = 0; fn < 2; ++fn)
      bfr[fn] = load_frag(&Bs[wn + fn * 16 + lrow][0], khalf);
#pragma unroll
    for (int fm = 0; fm < 2; ++fm)
#pragma unroll
      for (int fn = 0; fn < 2; ++fn)
        acc[fm][fn] = __builtin_amdgcn_wmma_f32_16x16x32_bf16(
            false, afr[fm], false, bfr[fn], (short)0, acc[fm][fn], false, false);
    __syncthreads();
  }

#pragma unroll
  for (int fm = 0; fm < 2; ++fm)
#pragma unroll
    for (int fn = 0; fn < 2; ++fn)
#pragma unroll
      for (int e = 0; e < 8; ++e) {
        int ti = wm + fm * 16 + e + ((lane >> 4) << 3);
        int si = wn + fn * 16 + (lane & 15);
        atomicAdd(&diag[ti - si + 63], acc[fm][fn][e]);
      }
  __syncthreads();

  for (int i = tid; i < 127; i += 128) {
    int tau = t0 - s0 + (i - 63);
    tau %= L; if (tau < 0) tau += L;
    atomicAdd(&corr[(size_t)b * L + tau], diag[i] * (1.0f / 256.0f));
  }
}

// ---------------------------------------------------------------------------
// Small helper kernels
// ---------------------------------------------------------------------------
__global__ void af_zero(float* __restrict__ p, int n) {
  int i = blockIdx.x * blockDim.x + threadIdx.x;
  if (i < n) p[i] = 0.f;
}

__global__ void af_add(const float* __restrict__ a, const float* __restrict__ b,
                       float* __restrict__ c, int n) {
  int i = blockIdx.x * blockDim.x + threadIdx.x;
  if (i < n) c[i] = a[i] + b[i];
}

// series_decomp: moving average k=25 with edge replication.
// tmode: 0 = seasonal only, 1 = also store trend, 2 = accumulate trend.
__global__ void af_decomp(const float* __restrict__ x, float* __restrict__ seasonal,
                          float* __restrict__ trendOut, int tmode,
                          int Bn, int L, int D) {
  int idx = blockIdx.x * blockDim.x + threadIdx.x;
  int total = Bn * L * D;
  if (idx >= total) return;
  int d = idx % D;
  int r = idx / D;
  int t = r % L;
  int b = r / L;
  const float* xb = x + (size_t)b * L * D + d;
  float s = 0.f;
  for (int j = -12; j <= 12; ++j) {
    int tt = t + j;
    tt = tt < 0 ? 0 : (tt >= L ? L - 1 : tt);
    s += xb[(size_t)tt * D];
  }
  float m = s * (1.0f / 25.0f);
  seasonal[idx] = x[idx] - m;
  if (tmode == 1) trendOut[idx] = m;
  else if (tmode == 2) trendOut[idx] += m;
}

// X[b,t,:] += mark[b,t,:] @ mw  (MD=4 tiny K, not worth WMMA)
__global__ void af_mark_embed(float* __restrict__ X, const float* __restrict__ mark,
                              const float* __restrict__ mw, int total, int D, int MD) {
  int idx = blockIdx.x * blockDim.x + threadIdx.x;
  if (idx >= total) return;
  int d = idx % D;
  int r = idx / D;
  float s = 0.f;
  for (int j = 0; j < MD; ++j) s += mark[(size_t)r * MD + j] * mw[(size_t)j * D + d];
  X[idx] += s;
}

// per-batch top-k (k<=8) + softmax over mean autocorrelation
__global__ void af_topk(float* __restrict__ corr, int L, int k,
                        int* __restrict__ delays, float* __restrict__ weights) {
  int b = blockIdx.x;
  int tid = threadIdx.x;
  __shared__ float bv[256];
  __shared__ int   bi[256];
  __shared__ float selv[8];
  __shared__ int   seli[8];
  float* c = corr + (size_t)b * L;
  for (int it = 0; it < k; ++it) {
    float best = -3.4e38f; int besti = 0;
    for (int t = tid; t < L; t += 256) {
      float v = c[t];
      if (v > best) { best = v; besti = t; }
    }
    bv[tid] = best; bi[tid] = besti;
    __syncthreads();
    for (int s = 128; s > 0; s >>= 1) {
      if (tid < s && bv[tid + s] > bv[tid]) { bv[tid] = bv[tid + s]; bi[tid] = bi[tid + s]; }
      __syncthreads();
    }
    if (tid == 0) { selv[it] = bv[0]; seli[it] = bi[0]; c[bi[0]] = -3.4e38f; }
    __syncthreads();
  }
  if (tid == 0) {
    float mx = selv[0];
    for (int i = 1; i < k; ++i) mx = fmaxf(mx, selv[i]);
    float e[8], sum = 0.f;
    for (int i = 0; i < k; ++i) { e[i] = expf(selv[i] - mx); sum += e[i]; }
    for (int i = 0; i < k; ++i) {
      weights[b * 8 + i] = e[i] / sum;
      delays[b * 8 + i]  = seli[i];
    }
  }
}

// agg[b,t,d] = sum_i w[b,i] * V[b, (t+delay[b,i]) % L, d]
__global__ void af_agg(const float* __restrict__ V, float* __restrict__ out,
                       const int* __restrict__ delays, const float* __restrict__ w,
                       int Bn, int L, int D, int k, int vBatchRows) {
  int idx = blockIdx.x * blockDim.x + threadIdx.x;
  int total = Bn * L * D;
  if (idx >= total) return;
  int d = idx % D;
  int r = idx / D;
  int t = r % L;
  int b = r / L;
  float s = 0.f;
  for (int i = 0; i < k; ++i) {
    int tt = t + delays[b * 8 + i];
    if (tt >= L) tt -= L;
    s += w[b * 8 + i] * V[((size_t)b * vBatchRows + tt) * D + d];
  }
  out[idx] = s;
}

// LayerNorm over last dim (D==256, blockDim==256), y = (x-mu)/sqrt(var+eps)*g+b
__global__ void af_ln(const float* __restrict__ X, float* __restrict__ Y,
                      const float* __restrict__ g, const float* __restrict__ beta,
                      int D) {
  int row = blockIdx.x;
  int d = threadIdx.x;
  __shared__ float red[256];
  float v = X[(size_t)row * D + d];
  red[d] = v;
  __syncthreads();
  for (int s = 128; s > 0; s >>= 1) { if (d < s) red[d] += red[d + s]; __syncthreads(); }
  float mu = red[0] / D;
  __syncthreads();
  float dv = v - mu;
  red[d] = dv * dv;
  __syncthreads();
  for (int s = 128; s > 0; s >>= 1) { if (d < s) red[d] += red[d + s]; __syncthreads(); }
  float var = red[0] / D;
  Y[(size_t)row * D + d] = dv * rsqrtf(var + 1e-5f) * g[d] + beta[d];
}

// subtract per-(b,d) time mean (axis=1), in place
__global__ void af_sub_time_mean(float* __restrict__ X, int Bn, int L, int D) {
  int idx = blockIdx.x * blockDim.x + threadIdx.x;
  if (idx >= Bn * D) return;
  int d = idx % D;
  int b = idx / D;
  float* p = X + (size_t)b * L * D + d;
  float s = 0.f;
  for (int t = 0; t < L; ++t) s += p[(size_t)t * D];
  float m = s / L;
  for (int t = 0; t < L; ++t) p[(size_t)t * D] -= m;
}

// mean over time of x(B,L,C) -> out(B,C)
__global__ void af_col_mean(const float* __restrict__ x, float* __restrict__ out,
                            int Bn, int L, int C) {
  int idx = blockIdx.x * blockDim.x + threadIdx.x;
  if (idx >= Bn * C) return;
  int c = idx % C;
  int b = idx / C;
  float s = 0.f;
  for (int t = 0; t < L; ++t) s += x[((size_t)b * L + t) * C + c];
  out[idx] = s / L;
}

// decoder inits: seasonal (first 512 rows then zeros), trend (first 512 then mean)
__global__ void af_dec_init(const float* __restrict__ seas, const float* __restrict__ trend,
                            const float* __restrict__ xmean,
                            float* __restrict__ S, float* __restrict__ T,
                            int Bn, int LD, int LABEL, int LE, int C) {
  int idx = blockIdx.x * blockDim.x + threadIdx.x;
  int total = Bn * LD * C;
  if (idx >= total) return;
  int c = idx % C;
  int r = idx / C;
  int t = r % LD;
  int b = r / LD;
  if (t < LABEL) {
    S[idx] = seas[((size_t)b * LE + t) * C + c];
    T[idx] = trend[((size_t)b * LE + t) * C + c];
  } else {
    S[idx] = 0.f;
    T[idx] = xmean[b * C + c];
  }
}

// maxpool window=3 stride=2 with -inf pads (conv_layer)
__global__ void af_maxpool(const float* __restrict__ Y, float* __restrict__ out,
                           int Bn, int Lin, int Lout, int D) {
  int idx = blockIdx.x * blockDim.x + threadIdx.x;
  int total = Bn * Lout * D;
  if (idx >= total) return;
  int d = idx % D;
  int r = idx / D;
  int to = r % Lout;
  int b = r / Lout;
  float m = -3.4e38f;
  for (int dt = -1; dt <= 1; ++dt) {
    int t = 2 * to + dt;
    if (t >= 0 && t < Lin) m = fmaxf(m, Y[((size_t)b * Lin + t) * D + d]);
  }
  out[idx] = m;
}

// out[b,t] = sum_d (T+S)[b,t,d]*cw[d] + cb
__global__ void af_final(const float* __restrict__ T, const float* __restrict__ S,
                         const float* __restrict__ cw, const float* __restrict__ cb,
                         float* __restrict__ out, int rows, int C) {
  int idx = blockIdx.x * blockDim.x + threadIdx.x;
  if (idx >= rows) return;
  float s = cb[0];
  for (int d = 0; d < C; ++d) s += (T[(size_t)idx * C + d] + S[(size_t)idx * C + d]) * cw[d];
  out[idx] = s;
}

// ---------------------------------------------------------------------------
// Host orchestration
// ---------------------------------------------------------------------------
static inline int cdiv(int a, int b) { return (a + b - 1) / b; }

struct AttnW { const float *wq, *bq, *wk, *bk, *wv, *bv, *wo, *bo; };

extern "C" void kernel_launch(void* const* d_in, const int* in_sizes, int n_in,
                              void* d_out, int out_size, void* d_ws, size_t ws_size,
                              hipStream_t stream) {
  (void)in_sizes; (void)n_in; (void)out_size; (void)ws_size;
  const int Bn = 32, LE = 3072, LE2 = 1536, LD = 1024, D = 256, FF = 1024;
  const int CIN = 64, CO = 64, MD = 4, LABEL = 512;

  const float* x_enc      = (const float*)d_in[0];
  const float* x_mark_enc = (const float*)d_in[1];
  const float* x_mark_dec = (const float*)d_in[3];
  auto P = [&](int i) { return (const float*)d_in[i]; };

  AttnW ea0{P(8),  P(9),  P(10), P(11), P(12), P(13), P(14), P(15)};
  const float *e0w1 = P(16), *e0w2 = P(17);
  AttnW ea1{P(18), P(19), P(20), P(21), P(22), P(23), P(24), P(25)};
  const float *e1w1 = P(26), *e1w2 = P(27);
  const float *convw = P(28), *convb = P(29);
  const float *normG = P(30), *normB = P(31);
  AttnW dsel{P(32), P(33), P(34), P(35), P(36), P(37), P(38), P(39)};
  AttnW dcro{P(40), P(41), P(42), P(43), P(44), P(45), P(46), P(47)};
  const float *dw1 = P(48), *dw2 = P(49), *trw = P(50);
  const float *projw = P(51), *projb = P(52), *compw = P(53), *compb = P(54);

  // -------- workspace carve-out (fp32) --------
  float* ws = (float*)d_ws;
  const size_t SZ = (size_t)Bn * LE * D;
  float* Xb    = ws; ws += SZ;
  float* Bb    = ws; ws += SZ;
  float* Cc    = ws; ws += SZ;
  float* Qb    = ws; ws += SZ;
  float* Kb    = ws; ws += SZ;
  float* Vb    = ws; ws += SZ;
  float* Hh    = ws; ws += (size_t)Bn * LE * FF;
  float* Ee    = ws; ws += (size_t)Bn * LE2 * D;
  float* Tacc  = ws; ws += (size_t)Bn * LD * D;
  float* seasE = ws; ws += (size_t)Bn * LE * CIN;
  float* trenE = ws; ws += (size_t)Bn * LE * CIN;
  float* Sini  = ws; ws += (size_t)Bn * LD * CIN;
  float* Ttr   = ws; ws += (size_t)Bn * LD * CIN;
  float* RT    = ws; ws += (size_t)Bn * LD * CO;
  float* xmean = ws; ws += (size_t)Bn * CIN;
  float* corr  = ws; ws += (size_t)Bn * LE;
  float* wts   = ws; ws += (size_t)Bn * 8;
  int*   dly   = (int*)ws;

  auto gemm = [&](const float* A, const float* W, const float* bias, float* C,
                  int M, int N, int K, int act, int amode, int L, int Cin) {
    dim3 g(cdiv(N, 64), cdiv(M, 64));
    gemm_wmma<<<g, 128, 0, stream>>>(A, W, bias, C, M, N, K, act, amode, L, Cin);
  };
  auto zero = [&](float* p, int n) { af_zero<<<cdiv(n, 256), 256, 0, stream>>>(p, n); };
  auto add  = [&](const float* a, const float* b, float* c, int n) {
    af_add<<<cdiv(n, 256), 256, 0, stream>>>(a, b, c, n);
  };
  auto decomp = [&](const float* x, float* s, float* t, int tmode, int L, int Dd) {
    int n = Bn * L * Dd;
    af_decomp<<<cdiv(n, 256), 256, 0, stream>>>(x, s, t, tmode, Bn, L, Dd);
  };
  auto dcopy = [&](float* dst, const float* src, int n) {
    hipMemcpyAsync(dst, src, (size_t)n * sizeof(float), hipMemcpyDeviceToDevice, stream);
  };

  // Full autocorrelation attention: OUT = attn(xq; xkv) (L = query length)
  auto attention = [&](const float* xq, int Lq, const float* xkv, int LkvRows,
                       const AttnW& a, int topk, float* AGG, float* OUT) {
    gemm(xq,  a.wq, a.bq, Qb, Bn * Lq,      D, D, 0, 0, 0, 0);
    gemm(xkv, a.wk, a.bk, Kb, Bn * LkvRows, D, D, 0, 0, 0, 0);
    gemm(xkv, a.wv, a.bv, Vb, Bn * LkvRows, D, D, 0, 0, 0, 0);
    zero(corr, Bn * Lq);
    dim3 cg(Lq / 64, Lq / 64, Bn);
    corr_wmma<<<cg, 128, 0, stream>>>(Qb, Kb, corr, Lq, D, Lq, LkvRows);
    af_topk<<<Bn, 256, 0, stream>>>(corr, Lq, topk, dly, wts);
    int n = Bn * Lq * D;
    af_agg<<<cdiv(n, 256), 256, 0, stream>>>(Vb, AGG, dly, wts, Bn, Lq, D, topk, LkvRows);
    gemm(AGG, a.wo, a.bo, OUT, Bn * Lq, D, D, 0, 0, 0, 0);
  };

  auto enc_layer = [&](float* X, int L, const AttnW& a, const float* w1,
                       const float* w2, int tk) {
    int n = Bn * L * D;
    attention(X, L, X, L, a, tk, Bb, Cc);
    add(X, Cc, X, n);
    decomp(X, Bb, nullptr, 0, L, D);              // Bb = seasonal
    gemm(Bb, w1, nullptr, Hh, Bn * L, FF, D, 1, 0, 0, 0);   // GELU
    gemm(Hh, w2, nullptr, Cc, Bn * L, D, FF, 0, 0, 0, 0);
    add(Bb, Cc, X, n);
    decomp(X, Bb, nullptr, 0, L, D);
    dcopy(X, Bb, n);
  };

  // ===================== encoder =====================
  decomp(x_enc, seasE, trenE, 1, LE, CIN);
  af_col_mean<<<cdiv(Bn * CIN, 64), 64, 0, stream>>>(x_enc, xmean, Bn, LE, CIN);

  gemm(x_enc, P(4), nullptr, Xb, Bn * LE, D, 3 * CIN, 0, 1, LE, CIN);   // conv emb
  af_mark_embed<<<cdiv(Bn * LE * D, 256), 256, 0, stream>>>(
      Xb, x_mark_enc, P(5), Bn * LE * D, D, MD);

  enc_layer(Xb, LE, ea0, e0w1, e0w2, 8);          // int(ln(3072)) = 8

  gemm(Xb, convw, convb, Cc, Bn * LE, D, 3 * D, 2, 1, LE, D);           // ELU conv
  af_maxpool<<<cdiv(Bn * LE2 * D, 256), 256, 0, stream>>>(Cc, Xb, Bn, LE, LE2, D);

  enc_layer(Xb, LE2, ea1, e1w1, e1w2, 7);         // int(ln(1536)) = 7

  af_ln<<<Bn * LE2, 256, 0, stream>>>(Xb, Ee, normG, normB, D);
  af_sub_time_mean<<<cdiv(Bn * D, 64), 64, 0, stream>>>(Ee, Bn, LE2, D);

  // ===================== decoder =====================
  af_dec_init<<<cdiv(Bn * LD * CIN, 256), 256, 0, stream>>>(
      seasE, trenE, xmean, Sini, Ttr, Bn, LD, LABEL, LE, CIN);

  gemm(Sini, P(6), nullptr, Xb, Bn * LD, D, 3 * CIN, 0, 1, LD, CIN);    // conv emb
  af_mark_embed<<<cdiv(Bn * LD * D, 256), 256, 0, stream>>>(
      Xb, x_mark_dec, P(7), Bn * LD * D, D, MD);

  {
    int n = Bn * LD * D;
    zero(Tacc, n);
    // self attention (int(ln(1024)) = 6)
    attention(Xb, LD, Xb, LD, dsel, 6, Bb, Cc);
    add(Xb, Cc, Xb, n);
    decomp(Xb, Bb, Tacc, 2, LD, D);
    dcopy(Xb, Bb, n);
    // cross attention (k/v from encoder, first 1024 of 1536 rows per batch)
    attention(Xb, LD, Ee, LE2, dcro, 6, Bb, Cc);
    add(Xb, Cc, Xb, n);
    decomp(Xb, Bb, Tacc, 2, LD, D);
    dcopy(Xb, Bb, n);
    // FFN
    gemm(Xb, dw1, nullptr, Hh, Bn * LD, FF, D, 1, 0, 0, 0);
    gemm(Hh, dw2, nullptr, Cc, Bn * LD, D, FF, 0, 0, 0, 0);
    add(Xb, Cc, Xb, n);
    decomp(Xb, Bb, Tacc, 2, LD, D);
    dcopy(Xb, Bb, n);
    // residual trend projection (circular conv3, 256 -> 64)
    gemm(Tacc, trw, nullptr, RT, Bn * LD, CO, 3 * D, 0, 1, LD, D);
    add(Ttr, RT, Ttr, Bn * LD * CO);
  }

  // final norm + projections
  af_ln<<<Bn * LD, 256, 0, stream>>>(Xb, Cc, normG, normB, D);
  af_sub_time_mean<<<cdiv(Bn * D, 64), 64, 0, stream>>>(Cc, Bn, LD, D);
  gemm(Cc, projw, projb, Bb, Bn * LD, CO, D, 0, 0, 0, 0);               // seasonal
  af_final<<<cdiv(Bn * LD, 256), 256, 0, stream>>>(
      Ttr, Bb, compw, compb, (float*)d_out, Bn * LD, CO);
}
